// CWAF_Block_55190329754129
// MI455X (gfx1250) — matrix operand order repaired and checked by reading, not compiled, not run on hardware
//
#include <hip/hip_runtime.h>
#include <math.h>

#define BB 4
#define CC 128
#define CR 32
#define HH 96
#define WW 320
#define DD 24
#define NN (HH*WW)          // 30720 pixels per batch
#define TOT (BB*NN)         // 122880 pixels total

typedef __attribute__((ext_vector_type(16))) __bf16 v16bf;
typedef __attribute__((ext_vector_type(8)))  float  v8f;

__device__ __forceinline__ unsigned short f2bf(float f){
  unsigned int u = __float_as_uint(f);
  u += 0x7FFFu + ((u >> 16) & 1u);           // round-to-nearest-even
  return (unsigned short)(u >> 16);
}
__device__ __forceinline__ float bf2f(unsigned short h){
  return __uint_as_float(((unsigned int)h) << 16);
}
__device__ __forceinline__ int refl(int x, int n){
  x = x < 0 ? -x : x;
  x = x >= n ? 2*n - 2 - x : x;
  return x;
}
__device__ __forceinline__ v16bf ldsLoad16bf(const unsigned short* p){
  union { uint4 q[2]; v16bf v; } u;
  const uint4* s = (const uint4*)p;      // 16B-aligned
  u.q[0] = s[0];
  u.q[1] = s[1];
  return u.v;
}

// ---------------------------------------------------------------------------
// Kernel 1: q = l2norm(W_q x + b_q), k = l2norm(W_k x + b_k); stored [B,H,W,Cr]
// ---------------------------------------------------------------------------
__global__ void qk_kernel(const float* __restrict__ t_feat, const float* __restrict__ s_feat,
                          const float* __restrict__ qw, const float* __restrict__ qb,
                          const float* __restrict__ kw, const float* __restrict__ kb,
                          float* __restrict__ qout, float* __restrict__ kout){
  __shared__ float wq[CR*CC];
  __shared__ float wk[CR*CC];
  __shared__ float bq[CR], bk[CR];
  for (int i = threadIdx.x; i < CR*CC; i += blockDim.x){ wq[i] = qw[i]; wk[i] = kw[i]; }
  if (threadIdx.x < CR){ bq[threadIdx.x] = qb[threadIdx.x]; bk[threadIdx.x] = kb[threadIdx.x]; }
  __syncthreads();
  int gp = blockIdx.x*blockDim.x + threadIdx.x;   // global pixel
  if (gp >= TOT) return;
  int b = gp / NN, pix = gp % NN;
  const float* tf = t_feat + (size_t)b*CC*NN + pix;
  const float* sf = s_feat + (size_t)b*CC*NN + pix;
  float acc[CR];
  for (int pass = 0; pass < 2; ++pass){
    const float* src = pass ? sf : tf;
    const float* wl  = pass ? wk : wq;
    const float* bl  = pass ? bk : bq;
    #pragma unroll
    for (int r = 0; r < CR; ++r) acc[r] = bl[r];
    for (int c = 0; c < CC; ++c){               // coalesced: wave reads 128B per c
      float x = src[(size_t)c*NN];
      #pragma unroll
      for (int r = 0; r < CR; ++r) acc[r] = fmaf(wl[r*CC + c], x, acc[r]);
    }
    float n2 = 0.f;
    #pragma unroll
    for (int r = 0; r < CR; ++r) n2 = fmaf(acc[r], acc[r], n2);
    float inv = 1.0f / fmaxf(sqrtf(n2), 1e-12f);
    float* dst = (pass ? kout : qout) + (size_t)gp*CR;
    #pragma unroll
    for (int r = 0; r < CR; ++r) dst[r] = acc[r]*inv;
  }
}

// ---------------------------------------------------------------------------
// Kernel 2: plane-sweep warp + sim + softmax(D) + cost, one wave32 per pixel,
// lane = channel. Writes cost [B,H,W,Cr] and prob [B,D,H,W] into d_out.
// ---------------------------------------------------------------------------
__global__ void warp_kernel(const float* __restrict__ q, const float* __restrict__ k,
                            const float* __restrict__ directs, const int* __restrict__ image_shape,
                            float* __restrict__ cost, float* __restrict__ prob_out){
  int wave = (blockIdx.x*blockDim.x + threadIdx.x) >> 5;
  int lane = threadIdx.x & 31;
  if (wave >= TOT) return;
  int b = wave / NN, rem = wave % NN;
  int h = rem / WW, w = rem % WW;
  float rel_scale = 640.0f / (float)image_shape[3];
  float coef = rel_scale * directs[b] * (float)(WW - 1);
  float qv = q[(size_t)wave*CR + lane];
  const float* krow = k + ((size_t)b*NN + (size_t)h*WW)*CR;
  float g[DD], s[DD];
  float m = -1e30f;
  #pragma unroll
  for (int d = 0; d < DD; ++d){
    float disp = (0.3f/23.0f)*(float)d;
    float ix = fminf(fmaxf((float)w + disp*coef, 0.0f), (float)(WW - 1));
    int ix0 = (int)floorf(ix);
    int ix1 = min(ix0 + 1, WW - 1);
    float w1 = ix - (float)ix0;
    float g0 = krow[(size_t)ix0*CR + lane];
    float g1 = krow[(size_t)ix1*CR + lane];
    float gv = g0 + (g1 - g0)*w1;
    g[d] = gv;
    float p = qv*gv;
    #pragma unroll
    for (int off = 16; off > 0; off >>= 1) p += __shfl_xor(p, off, 32);
    s[d] = p;
    m = fmaxf(m, p);
  }
  float sum = 0.f;
  #pragma unroll
  for (int d = 0; d < DD; ++d){ s[d] = __expf(s[d] - m); sum += s[d]; }
  float inv = 1.0f/sum;
  float cv = 0.f;
  #pragma unroll
  for (int d = 0; d < DD; ++d){ s[d] *= inv; cv = fmaf(g[d], s[d], cv); }
  cost[(size_t)wave*CR + lane] = cv;
  #pragma unroll
  for (int d = 0; d < DD; ++d)
    if (lane == d)   // all lanes hold identical s[]; lane d writes plane d
      prob_out[(((size_t)b*DD + d)*HH + h)*WW + w] = s[d];
}

// ---------------------------------------------------------------------------
// Kernel 3: cost_exp (Cr->C), alpha gate, fused = a*t + (1-a)*ce;
// writes fused as bf16 hi/lo in [B,H,W,C] (channel-contiguous for WMMA B-tiles)
// ---------------------------------------------------------------------------
__global__ void fuse_kernel(const float* __restrict__ t_feat, const float* __restrict__ cost,
                            const float* __restrict__ ce_w, const float* __restrict__ ce_b,
                            const float* __restrict__ fa_w, const float* __restrict__ fa_b,
                            unsigned short* __restrict__ fusedHi, unsigned short* __restrict__ fusedLo){
  __shared__ float costL[32*CR];       // 32 pixels x 32 channels
  __shared__ float tL[CC*32];          // 128 channels x 32 pixels
  __shared__ float cwL[CC*CR];
  __shared__ float partial[256];
  __shared__ float alphaL[32];
  __shared__ unsigned short hiL[32*CC];
  __shared__ unsigned short loL[32*CC];
  int tile = blockIdx.x;
  int b = tile / (NN/32);
  int pix0 = (tile % (NN/32))*32;      // within batch
  int t = threadIdx.x;
  for (int i = t; i < 32*CR; i += 256) costL[i] = cost[((size_t)b*NN + pix0)*CR + i];
  {
    int p = t & 31;
    for (int i = 0; i < 16; ++i){
      int c = (t >> 5) + i*8;
      tL[c*32 + p] = t_feat[((size_t)b*CC + c)*NN + pix0 + p];
    }
  }
  for (int i = t; i < CC*CR; i += 256) cwL[i] = ce_w[i];
  __syncthreads();
  int p  = t & 31;
  int c0 = t >> 5;
  float ce[16];
  float pa = 0.f;
  #pragma unroll
  for (int i = 0; i < 16; ++i){
    int c = c0 + i*8;
    float a = ce_b[c];
    #pragma unroll
    for (int r = 0; r < CR; ++r) a = fmaf(cwL[c*CR + r], costL[p*CR + r], a);
    ce[i] = a;
    pa = fmaf(fa_w[c], tL[c*32 + p], pa);
    pa = fmaf(fa_w[CC + c], a, pa);
  }
  partial[t] = pa;
  __syncthreads();
  if (t < 32){
    float s = fa_b[0];
    #pragma unroll
    for (int j = 0; j < 8; ++j) s += partial[j*32 + t];
    alphaL[t] = 1.0f/(1.0f + __expf(-s));
  }
  __syncthreads();
  float al = alphaL[p];
  #pragma unroll
  for (int i = 0; i < 16; ++i){
    int c = c0 + i*8;
    float f = al*tL[c*32 + p] + (1.0f - al)*ce[i];
    unsigned short hi = f2bf(f);
    unsigned short lo = f2bf(f - bf2f(hi));
    hiL[p*CC + c] = hi;
    loL[p*CC + c] = lo;
  }
  __syncthreads();
  const unsigned int* hs = (const unsigned int*)hiL;
  const unsigned int* ls = (const unsigned int*)loL;
  unsigned int* hd = (unsigned int*)fusedHi + ((size_t)b*NN + pix0)*(CC/2);
  unsigned int* ld = (unsigned int*)fusedLo + ((size_t)b*NN + pix0)*(CC/2);
  for (int i = t; i < 32*CC/2; i += 256){ hd[i] = hs[i]; ld[i] = ls[i]; }
}

// ---------------------------------------------------------------------------
// Kernel 4: pack redu_w into per-lane A-matrix VGPR order (bf16 hi/lo split).
// tile = (tap*4 + cinblk)*8 + m0;  K = j + (lane>=16?16:0)
// ---------------------------------------------------------------------------
__global__ void pack_kernel(const float* __restrict__ rw,
                            unsigned short* __restrict__ Ahi, unsigned short* __restrict__ Alo){
  int e = blockIdx.x*blockDim.x + threadIdx.x;
  const int TOTE = 288*32*16;
  if (e >= TOTE) return;
  int j    = e & 15;
  int lane = (e >> 4) & 31;
  int tIdx = e >> 9;
  int m0 = tIdx & 7;
  int cb = (tIdx >> 3) & 3;
  int tap = tIdx >> 5;
  int ky = tap/3, kx = tap%3;
  int v = j >> 1;
  int kk = j + (lane >= 16 ? 16 : 0);
  int cout = m0*16 + (lane & 15);
  int cin  = cb*32 + kk;
  float wv = rw[(((size_t)cout*CC + cin)*3 + ky)*3 + kx];
  unsigned short hi = f2bf(wv);
  unsigned short lo = f2bf(wv - bf2f(hi));
  Ahi[e] = hi; Alo[e] = lo;
}

// ---------------------------------------------------------------------------
// Kernel 5: 3x3 reflect conv as implicit GEMM with split-bf16 WMMA.
// Halo tile async-staged into LDS once per block (ASYNCcnt); 6 rotating
// accumulators keep dependent WMMAs >5 slots apart (ISA 7.12.1 bf16 hazard),
// so the matrix pipe runs without the 1-NOP+4-coexec bubbles.
// ---------------------------------------------------------------------------
#define TCOLS 18
#define CSTRIDE 136   // ushorts per column (272 B = 68 dwords -> bank 4*l per lane)

__global__ void conv_kernel(const unsigned short* __restrict__ fHi, const unsigned short* __restrict__ fLo,
                            const unsigned short* __restrict__ Ahi, const unsigned short* __restrict__ Alo,
                            float* __restrict__ convbuf){
  __shared__ unsigned short tileHi[3*TCOLS*CSTRIDE];
  __shared__ unsigned short tileLo[3*TCOLS*CSTRIDE];
  int blk = blockIdx.x;
  int wt = blk % (WW/16);
  int h  = (blk / (WW/16)) % HH;
  int b  = blk / ((WW/16)*HH);
  int w0 = wt*16;
  int tid = threadIdx.x;

  // ---- async stage: 2 bufs x 3 rows x 18 cols x 16 segs of 16B = 1728 chunks
  for (int qd = tid; qd < 1728; qd += 256){
    int buf  = qd / 864;
    int rem  = qd % 864;
    int row  = rem / 288;
    int rem2 = rem % 288;
    int col  = rem2 / 16;
    int seg  = rem2 & 15;
    int hh = refl(h + row - 1, HH);
    int ww = refl(w0 + col - 1, WW);
    const unsigned short* src = (buf ? fLo : fHi)
        + ((size_t)b*NN + (size_t)hh*WW + ww)*CC + seg*8;
    unsigned short* dst = (buf ? tileLo : tileHi) + (row*TCOLS + col)*CSTRIDE + seg*8;
    unsigned int ldsaddr = (unsigned int)(size_t)dst;           // addrspace(3) offset
    unsigned long long gaddr = (unsigned long long)(size_t)src;
    asm volatile("global_load_async_to_lds_b128 %0, %1, off"
                 :: "v"(ldsaddr), "v"(gaddr) : "memory");
  }
#if __has_builtin(__builtin_amdgcn_s_wait_asynccnt)
  __builtin_amdgcn_s_wait_asynccnt(0);
#else
  asm volatile("s_wait_asynccnt 0" ::: "memory");
#endif
  __syncthreads();

  int m0   = tid >> 5;
  int lane = tid & 31;
  int col  = lane & 15;
  int koff = (lane & 16) ? 16 : 0;
  v8f acc[6];
  #pragma unroll
  for (int i = 0; i < 6; ++i) acc[i] = (v8f){0.f,0.f,0.f,0.f,0.f,0.f,0.f,0.f};
  #pragma unroll
  for (int ky = 0; ky < 3; ++ky){
    #pragma unroll
    for (int kx = 0; kx < 3; ++kx){
      int tcol = (ky*TCOLS + col + kx)*CSTRIDE;
      int tap = ky*3 + kx;
      #pragma unroll
      for (int cb = 0; cb < 4; ++cb){
        v16bf Bh = ldsLoad16bf(tileHi + tcol + cb*32 + koff);
        v16bf Bl = ldsLoad16bf(tileLo + tcol + cb*32 + koff);
        size_t aoff = (((size_t)(tap*4 + cb)*8 + m0)*32 + lane)*16;
        v16bf Ah = *reinterpret_cast<const v16bf*>(Ahi + aoff);
        v16bf Al = *reinterpret_cast<const v16bf*>(Alo + aoff);
        // fp32 ~= hi*hi + hi*lo + lo*hi; rotate 6 accumulators so any
        // accumulator is reused only every 6 WMMAs (beyond the 5-slot hazard)
        int base = ((tap*4 + cb) & 1)*3;
        acc[base+0] = __builtin_amdgcn_wmma_f32_16x16x32_bf16(false, Ah, false, Bh, (short)0, acc[base+0], false, false);
        acc[base+1] = __builtin_amdgcn_wmma_f32_16x16x32_bf16(false, Ah, false, Bl, (short)0, acc[base+1], false, false);
        acc[base+2] = __builtin_amdgcn_wmma_f32_16x16x32_bf16(false, Al, false, Bh, (short)0, acc[base+2], false, false);
      }
    }
  }
  int coutBase = m0*16 + ((lane & 16) ? 8 : 0);
  size_t pix = (size_t)h*WW + w0 + col;
  #pragma unroll
  for (int r = 0; r < 8; ++r){
    float v = ((acc[0][r] + acc[1][r]) + (acc[2][r] + acc[3][r])) + (acc[4][r] + acc[5][r]);
    convbuf[((size_t)b*CC + coutBase + r)*NN + pix] = v;
  }
}

// ---------------------------------------------------------------------------
// Kernels 6-8: global mean pool, SE MLP, ELU gating into d_out
// ---------------------------------------------------------------------------
__global__ void pool_kernel(const float* __restrict__ convbuf, float* __restrict__ pooled){
  __shared__ float red[256];
  int row = blockIdx.x;                // b*128 + c
  const float* src = convbuf + (size_t)row*NN;
  float s = 0.f;
  for (int i = threadIdx.x; i < NN; i += 256) s += src[i];
  red[threadIdx.x] = s; __syncthreads();
  for (int o = 128; o > 0; o >>= 1){
    if (threadIdx.x < o) red[threadIdx.x] += red[threadIdx.x + o];
    __syncthreads();
  }
  if (threadIdx.x == 0) pooled[row] = red[0] / (float)NN;
}

__global__ void se_kernel(const float* __restrict__ pooled, const float* __restrict__ w1,
                          const float* __restrict__ w2, float* __restrict__ gate){
  __shared__ float pl[CC];
  __shared__ float hid[8];
  int b = blockIdx.x;
  int t = threadIdx.x;
  pl[t] = pooled[b*CC + t];
  __syncthreads();
  if (t < 8){
    float s = 0.f;
    for (int c = 0; c < CC; ++c) s = fmaf(w1[t*CC + c], pl[c], s);
    hid[t] = fmaxf(s, 0.f);
  }
  __syncthreads();
  float s = 0.f;
  #pragma unroll
  for (int r = 0; r < 8; ++r) s = fmaf(w2[t*8 + r], hid[r], s);
  gate[b*CC + t] = 1.0f/(1.0f + __expf(-s));
}

__global__ void out_kernel(const float* __restrict__ convbuf, const float* __restrict__ gate,
                           float* __restrict__ out){
  size_t idx = (size_t)blockIdx.x*blockDim.x + threadIdx.x;
  if (idx >= (size_t)BB*CC*NN) return;
  size_t bc = idx / NN;
  float x = convbuf[idx]*gate[bc];
  out[idx] = x > 0.f ? x : (__expf(x) - 1.0f);   // ELU(alpha=1)
}

// ---------------------------------------------------------------------------
extern "C" void kernel_launch(void* const* d_in, const int* in_sizes, int n_in,
                              void* d_out, int out_size, void* d_ws, size_t ws_size,
                              hipStream_t stream){
  const float* t_feat      = (const float*)d_in[0];
  const float* s_feat      = (const float*)d_in[1];
  const float* directs     = (const float*)d_in[2];
  const int*   image_shape = (const int*)  d_in[3];
  const float* q_w  = (const float*)d_in[4];
  const float* q_b  = (const float*)d_in[5];
  const float* k_w  = (const float*)d_in[6];
  const float* k_b  = (const float*)d_in[7];
  const float* ce_w = (const float*)d_in[8];
  const float* ce_b = (const float*)d_in[9];
  const float* fa_w = (const float*)d_in[10];
  const float* fa_b = (const float*)d_in[11];
  const float* redu_w = (const float*)d_in[12];
  const float* se_w1  = (const float*)d_in[13];
  const float* se_w2  = (const float*)d_in[14];

  char* ws = (char*)d_ws;
  size_t off = 0;
  float* q    = (float*)(ws + off); off += (size_t)TOT*CR*4;
  float* kbuf = (float*)(ws + off); off += (size_t)TOT*CR*4;
  float* cost = (float*)(ws + off); off += (size_t)TOT*CR*4;
  unsigned short* fHi = (unsigned short*)(ws + off); off += (size_t)TOT*CC*2;
  unsigned short* fLo = (unsigned short*)(ws + off); off += (size_t)TOT*CC*2;
  float* convbuf = (float*)(ws + off); off += (size_t)TOT*CC*4;
  unsigned short* Ahi = (unsigned short*)(ws + off); off += 288*512*2;
  unsigned short* Alo = (unsigned short*)(ws + off); off += 288*512*2;
  float* pooled = (float*)(ws + off); off += BB*CC*4;
  float* gate   = (float*)(ws + off); off += BB*CC*4;

  float* outMain = (float*)d_out;
  float* probOut = outMain + (size_t)BB*CC*NN;

  pack_kernel<<<(288*512 + 255)/256, 256, 0, stream>>>(redu_w, Ahi, Alo);
  qk_kernel  <<<TOT/256, 256, 0, stream>>>(t_feat, s_feat, q_w, q_b, k_w, k_b, q, kbuf);
  warp_kernel<<<TOT/8,   256, 0, stream>>>(q, kbuf, directs, image_shape, cost, probOut);
  fuse_kernel<<<TOT/32,  256, 0, stream>>>(t_feat, cost, ce_w, ce_b, fa_w, fa_b, fHi, fLo);
  conv_kernel<<<BB*HH*(WW/16), 256, 0, stream>>>(fHi, fLo, Ahi, Alo, convbuf);
  pool_kernel<<<BB*CC, 256, 0, stream>>>(convbuf, pooled);
  se_kernel  <<<BB, CC, 0, stream>>>(pooled, se_w1, se_w2, gate);
  out_kernel <<<(BB*CC*NN + 255)/256, 256, 0, stream>>>(convbuf, gate, outMain);
}
